// Attention_44074954391673
// MI455X (gfx1250) — matrix operand order repaired and checked by reading, not compiled, not run on hardware
//
#include <hip/hip_runtime.h>
#include <stdint.h>

// ---------------------------------------------------------------------------
// MI455X (gfx1250) attention pipeline, bf16 WMMA (v_wmma_f32_16x16x32_bf16)
// + async global->LDS tile movement (GLOBAL_LOAD_ASYNC_TO_LDS_B128, ASYNCcnt).
// Roofline: ~37 GFLOP math vs ~1 GB traffic @ 23.3 TB/s -> ~45us memory floor;
// bf16 WMMA (16 KFLOP/inst) keeps compute off the critical path.
// __launch_bounds__(256,2) raises the VGPR budget (8 waves / 4 SIMDs is
// already 2 waves/EU) so accumulators are not spilled to scratch.
// ---------------------------------------------------------------------------

typedef __attribute__((ext_vector_type(16))) __bf16 v16bf;
typedef __attribute__((ext_vector_type(8)))  float  v8f;
typedef __attribute__((ext_vector_type(4)))  int    v4i;
typedef unsigned short u16;

#define LSTR 72  // LDS row stride in halves: 144B (16B aligned, bank-friendly)

#if __has_builtin(__builtin_amdgcn_global_load_async_to_lds_b128) && \
    __has_builtin(__builtin_amdgcn_s_wait_asynccnt)
#define ASYNC_LDS 1
#else
#define ASYNC_LDS 0
#endif

__device__ __forceinline__ void copy16_g2l(u16* dst, const u16* src) {
#if ASYNC_LDS
  // signature (from hipcc diagnostic): (v4i addrspace(1)*, v4i addrspace(3)*,
  // imm offset, imm cpol)
  __builtin_amdgcn_global_load_async_to_lds_b128(
      (__attribute__((address_space(1))) v4i*)src,
      (__attribute__((address_space(3))) v4i*)dst, 0, 0);
#else
  *(uint4*)dst = *(const uint4*)src;
#endif
}

__device__ __forceinline__ void async_join() {
#if ASYNC_LDS
  __builtin_amdgcn_s_wait_asynccnt(0);
#endif
}

__device__ __forceinline__ u16 f2bf(float x) {
  union { float f; unsigned u; } c; c.f = x;
  unsigned u = c.u;
  unsigned r = (u + 0x7FFFu + ((u >> 16) & 1u)) >> 16;  // round-to-nearest-even
  return (u16)r;
}

// A-matrix frag (16x32 bf16), ISA layout: lanes 0-15 row m hold K{0-7,16-23},
// lanes 16-31 hold K{8-15,24-31}.
__device__ __forceinline__ v16bf load_a_frag(const u16* lds, int row0, int kh) {
  unsigned lane = threadIdx.x & 31u;
  unsigned sel  = lane >> 4;
  unsigned m    = lane & 15u;
  const u16* r = lds + (size_t)(row0 + (int)m) * LSTR + kh * 32;
  union { v16bf v; uint4 q[2]; } f;
  f.q[0] = *(const uint4*)(r + 8 * sel);
  f.q[1] = *(const uint4*)(r + 16 + 8 * sel);
  return f.v;
}

// B-matrix frag (32x16 bf16), ISA layout: lanes 0-15 col n hold K{0-7,16-23},
// lanes 16-31 hold K{8-15,24-31}. Column n = LDS row (row0+n).
__device__ __forceinline__ v16bf load_b_frag(const u16* lds, int row0, int kh) {
  unsigned lane = threadIdx.x & 31u;
  unsigned sel  = lane >> 4;
  unsigned n    = lane & 15u;
  const u16* r = lds + (size_t)(row0 + (int)n) * LSTR + kh * 32;
  union { v16bf v; uint4 q[2]; } f;
  f.q[0] = *(const uint4*)(r + 8 * sel);
  f.q[1] = *(const uint4*)(r + 16 + 8 * sel);
  return f.v;
}

__device__ __forceinline__ v8f wmma_bf16(v16bf a, v16bf b, v8f c) {
  return __builtin_amdgcn_wmma_f32_16x16x32_bf16(false, a, false, b, (short)0,
                                                 c, false, false);
}

// cooperative copy: rows x 64-half tile, global(sstride halves) -> LDS(LSTR)
__device__ __forceinline__ void load_tile64(u16* dst, const u16* src,
                                            size_t sstride, int rows) {
  int tid = threadIdx.x;
  int total = rows * 8;  // 8 x 16B chunks per 64-half row
  for (int idx = tid; idx < total; idx += 256) {
    int rr = idx >> 3, c = idx & 7;
    copy16_g2l(dst + (size_t)rr * LSTR + c * 8,
               src + (size_t)rr * sstride + c * 8);
  }
}

// --------------------------- conversion kernels ----------------------------

__global__ void conv_f2b(const float* __restrict__ in, u16* __restrict__ out, int n) {
  int i = blockIdx.x * 256 + threadIdx.x;
  if (i < n) out[i] = f2bf(in[i]);
}

__global__ void conv_transpose(const float* __restrict__ in, u16* __restrict__ out,
                               int R, int C) {
  int i = blockIdx.x * 256 + threadIdx.x;
  if (i < R * C) {
    int r = i / C, c = i % C;
    out[(size_t)c * R + r] = f2bf(in[i]);
  }
}

// ssan_w [5,12,64(d),64(e)] -> ssanT [5,12,64(e),64(d)]
__global__ void conv_ssanT(const float* __restrict__ in, u16* __restrict__ out) {
  int i = blockIdx.x * 256 + threadIdx.x;
  if (i < 5 * 12 * 64 * 64) {
    int e = i & 63, d = (i >> 6) & 63, ih = i >> 12;
    out[((size_t)ih * 64 + e) * 64 + d] = f2bf(in[i]);
  }
}

// ------------------------------- QKV GEMM ----------------------------------
// C[2048 x 768] = X[2048 x 768] @ W (B-operand from W^T rows). Output scattered
// head-major bf16 [B,H,L,64]; for V also a transposed copy [B,H,64,L].
__global__ void __launch_bounds__(256, 2)
gemm_qkv(const u16* __restrict__ X, const u16* __restrict__ WT,
         const float* __restrict__ bias,
         u16* __restrict__ outHM, u16* __restrict__ outT, int writeT) {
  __shared__ u16 sa[128 * LSTR];
  __shared__ u16 sb[128 * LSTR];
  const int m0 = blockIdx.x * 128;
  const int n0 = blockIdx.y * 128;
  const int tid = threadIdx.x;
  const int wave = tid >> 5, wr = wave >> 1, wc = wave & 1;
  const unsigned lane = tid & 31u, sel = lane >> 4, jl = lane & 15u;

  v8f zero = {};
  v8f acc[2][4];
#pragma unroll
  for (int a = 0; a < 2; ++a)
#pragma unroll
    for (int b = 0; b < 4; ++b) acc[a][b] = zero;

  for (int k0 = 0; k0 < 768; k0 += 64) {
    load_tile64(sa, X + (size_t)m0 * 768 + k0, 768, 128);
    load_tile64(sb, WT + (size_t)n0 * 768 + k0, 768, 128);
    async_join();
    __syncthreads();
#pragma unroll
    for (int t2 = 0; t2 < 2; ++t2) {
      int i0 = wr * 32 + t2 * 16;
      v16bf a0 = load_a_frag(sa, i0, 0);
      v16bf a1 = load_a_frag(sa, i0, 1);
#pragma unroll
      for (int t = 0; t < 4; ++t) {
        int j0 = wc * 64 + t * 16;
        v16bf b0 = load_b_frag(sb, j0, 0);
        v16bf b1 = load_b_frag(sb, j0, 1);
        acc[t2][t] = wmma_bf16(a0, b0, acc[t2][t]);
        acc[t2][t] = wmma_bf16(a1, b1, acc[t2][t]);
      }
    }
    __syncthreads();
  }
#pragma unroll
  for (int t2 = 0; t2 < 2; ++t2)
#pragma unroll
    for (int t = 0; t < 4; ++t)
#pragma unroll
      for (int v = 0; v < 8; ++v) {
        int m = m0 + wr * 32 + t2 * 16 + v + (int)sel * 8;
        int n = n0 + wc * 64 + t * 16 + (int)jl;
        float val = acc[t2][t][v] + bias[n];
        int b = m >> 10, l = m & 1023, hh = n >> 6, dh = n & 63;
        u16 bvv = f2bf(val);
        outHM[(((size_t)(b * 12 + hh)) * 1024 + l) * 64 + dh] = bvv;
        if (writeT)
          outT[(((size_t)(b * 12 + hh)) * 64 + dh) * 1024 + l] = bvv;
      }
}

// ------------------------------- qw GEMM -----------------------------------
// qw[i,b,h] = q[b,h] (1024x64) @ ssanT[i,h] (64x64).  grid=(L/128, 5*24)
__global__ void __launch_bounds__(256, 2)
gemm_qw(const u16* __restrict__ Q, const u16* __restrict__ SSW,
        u16* __restrict__ QW) {
  __shared__ u16 sa[128 * LSTR];
  __shared__ u16 sb[64 * LSTR];
  const int l0 = blockIdx.x * 128;
  const int i5 = blockIdx.y / 24;
  const int bh = blockIdx.y % 24;
  const int hh = bh % 12;
  const int tid = threadIdx.x;
  const int wave = tid >> 5;
  const unsigned lane = tid & 31u, sel = lane >> 4, jl = lane & 15u;

  load_tile64(sa, Q + ((size_t)bh * 1024 + l0) * 64, 64, 128);
  load_tile64(sb, SSW + ((size_t)(i5 * 12 + hh)) * 64 * 64, 64, 64);
  async_join();
  __syncthreads();

  const int i0 = wave * 16;
  v16bf a0 = load_a_frag(sa, i0, 0);
  v16bf a1 = load_a_frag(sa, i0, 1);
#pragma unroll
  for (int t = 0; t < 4; ++t) {
    v16bf b0 = load_b_frag(sb, t * 16, 0);
    v16bf b1 = load_b_frag(sb, t * 16, 1);
    v8f acc = {};
    acc = wmma_bf16(a0, b0, acc);
    acc = wmma_bf16(a1, b1, acc);
#pragma unroll
    for (int v = 0; v < 8; ++v) {
      int l = l0 + i0 + v + (int)sel * 8;
      int n = t * 16 + (int)jl;
      QW[(((size_t)i5 * 24 + bh) * 1024 + l) * 64 + n] = f2bf(acc[v]);
    }
  }
}

// ------------------------------ scores kernel ------------------------------
// Per block: one (b,h), 128x128 score tile. 8 waves x 8 subtiles x 20 WMMA.
// scores = (q.k + q.e[l-r] + k.e[l-r])/8 + mask + sum_i (qw_i.k + ab_i)*S_i
__global__ void __launch_bounds__(256, 2)
scores_kernel(const u16* __restrict__ Q, const u16* __restrict__ K,
              const u16* __restrict__ QW, const u16* __restrict__ E,
              const float* __restrict__ SM, const float* __restrict__ MASK,
              const float* __restrict__ AB, float* __restrict__ S) {
  extern __shared__ u16 smem[];
  u16* sq = smem;                  // 128*LSTR
  u16* sk = sq + 128 * LSTR;       // 128*LSTR
  u16* se = sk + 128 * LSTR;       // 256*LSTR (distance-embedding window)
  u16* sw = se + 256 * LSTR;       // 5*128*LSTR
  float* sg = (float*)(sw + 5 * 128 * LSTR);  // 8 waves * 2 * 16*33 f32

  const int bh = blockIdx.y;
  const int b = bh / 12, hh = bh % 12;
  const int qt = blockIdx.x >> 3, kt = blockIdx.x & 7;
  const int l0 = qt * 128, r0 = kt * 128;
  const int D0 = l0 - r0 + 896;  // window base distance index (>= 0)
  const int tid = threadIdx.x;
  const int wave = tid >> 5, wr = wave >> 1, wc = wave & 1;
  const unsigned lane = tid & 31u, sel = lane >> 4, jl = lane & 15u;

  load_tile64(sq, Q + ((size_t)bh * 1024 + l0) * 64, 64, 128);
  load_tile64(sk, K + ((size_t)bh * 1024 + r0) * 64, 64, 128);
  for (int i5 = 0; i5 < 5; ++i5)
    load_tile64(sw + i5 * 128 * LSTR,
                QW + (((size_t)i5 * 24 + bh) * 1024 + l0) * 64, 64, 128);
  for (int idx = tid; idx < 256 * 8; idx += 256) {
    int rr = idx >> 3, c = idx & 7;
    int gd = D0 + rr; if (gd > 2046) gd = 2046;
    copy16_g2l(se + (size_t)rr * LSTR + c * 8, E + (size_t)gd * 64 + c * 8);
  }
  async_join();
  __syncthreads();

  float ab[5];
#pragma unroll
  for (int i5 = 0; i5 < 5; ++i5) ab[i5] = AB[i5 * 12 + hh];

  float* lgq = sg + wave * (2 * 16 * 33);
  float* lgk = lgq + 16 * 33;

  for (int t2 = 0; t2 < 2; ++t2) {
    const int i0 = wr * 32 + t2 * 16;
    v16bf qa0 = load_a_frag(sq, i0, 0);
    v16bf qa1 = load_a_frag(sq, i0, 1);
    for (int t = 0; t < 4; ++t) {
      const int j0 = wc * 64 + t * 16;
      const int sbase = i0 - j0 + 112;  // in [0,224]; uses rows sbase..sbase+31

      // positional GEMM-lets (operands transient; results parked in LDS)
      {
        v16bf ka0 = load_a_frag(sk, j0, 0);
        v16bf ka1 = load_a_frag(sk, j0, 1);
        v16bf e00 = load_b_frag(se, sbase, 0);
        v16bf e01 = load_b_frag(se, sbase, 1);
        v16bf e10 = load_b_frag(se, sbase + 16, 0);
        v16bf e11 = load_b_frag(se, sbase + 16, 1);
        v8f gq0 = {}, gq1 = {}, gk0 = {}, gk1 = {};
        gq0 = wmma_bf16(qa0, e00, gq0); gq0 = wmma_bf16(qa1, e01, gq0);
        gq1 = wmma_bf16(qa0, e10, gq1); gq1 = wmma_bf16(qa1, e11, gq1);
        gk0 = wmma_bf16(ka0, e00, gk0); gk0 = wmma_bf16(ka1, e01, gk0);
        gk1 = wmma_bf16(ka0, e10, gk1); gk1 = wmma_bf16(ka1, e11, gk1);
#pragma unroll
        for (int v = 0; v < 8; ++v) {
          int gi = v + (int)sel * 8;
          lgq[gi * 33 + (int)jl]      = gq0[v];
          lgq[gi * 33 + 16 + (int)jl] = gq1[v];
          lgk[gi * 33 + (int)jl]      = gk0[v];
          lgk[gi * 33 + 16 + (int)jl] = gk1[v];
        }
      }

      v16bf kb0 = load_b_frag(sk, j0, 0);
      v16bf kb1 = load_b_frag(sk, j0, 1);

      const int r = r0 + j0 + (int)jl;
      const float mval = MASK[b * 1024 + r];

      float s[8];
      {
        v8f aqk = {};
        aqk = wmma_bf16(qa0, kb0, aqk);
        aqk = wmma_bf16(qa1, kb1, aqk);
#pragma unroll
        for (int v = 0; v < 8; ++v) {
          int i = v + (int)sel * 8;
          int c = 15 + i - (int)jl;  // diagonal gather index, in [0,30]
          s[v] = (aqk[v] + lgq[i * 33 + c] + lgk[(int)jl * 33 + c]) * 0.125f
                 + mval;
        }
      }

      // SSAN bilinear biases, one at a time (keeps ~1 accumulator live)
      for (int i5 = 0; i5 < 5; ++i5) {
        v16bf wa0 = load_a_frag(sw + i5 * 128 * LSTR, i0, 0);
        v16bf wa1 = load_a_frag(sw + i5 * 128 * LSTR, i0, 1);
        v8f aw = {};
        aw = wmma_bf16(wa0, kb0, aw);
        aw = wmma_bf16(wa1, kb1, aw);
        const float abi = ab[i5];
        const float* sp = SM + (size_t)i5 * 2097152 +
                          ((size_t)b * 1024) * 1024 + r;
#pragma unroll
        for (int v = 0; v < 8; ++v) {
          int l = l0 + i0 + v + (int)sel * 8;
          s[v] += (aw[v] + abi) * sp[(size_t)l * 1024];
        }
      }

#pragma unroll
      for (int v = 0; v < 8; ++v) {
        int l = l0 + i0 + v + (int)sel * 8;
        S[((size_t)bh * 1024 + l) * 1024 + r] = s[v];
      }
    }
  }
}

// ------------------------------- softmax -----------------------------------
__global__ void softmax_kernel(const float* __restrict__ S, u16* __restrict__ P) {
  __shared__ float red[256];
  const size_t row = blockIdx.x;
  const int tid = threadIdx.x;
  const float4 x = ((const float4*)(S + row * 1024))[tid];
  float m = fmaxf(fmaxf(x.x, x.y), fmaxf(x.z, x.w));
  red[tid] = m; __syncthreads();
  for (int s = 128; s > 0; s >>= 1) {
    if (tid < s) red[tid] = fmaxf(red[tid], red[tid + s]);
    __syncthreads();
  }
  const float mx = red[0]; __syncthreads();
  float e0 = __expf(x.x - mx), e1 = __expf(x.y - mx);
  float e2 = __expf(x.z - mx), e3 = __expf(x.w - mx);
  red[tid] = e0 + e1 + e2 + e3; __syncthreads();
  for (int s = 128; s > 0; s >>= 1) {
    if (tid < s) red[tid] += red[tid + s];
    __syncthreads();
  }
  const float inv = 1.0f / red[0];
  ushort4 o;
  o.x = f2bf(e0 * inv); o.y = f2bf(e1 * inv);
  o.z = f2bf(e2 * inv); o.w = f2bf(e3 * inv);
  ((ushort4*)(P + row * 1024))[tid] = o;
}

// -------------------------------- PV GEMM ----------------------------------
// ctx[b,h] (1024x64) = probs[b,h] (1024x1024) @ v[b,h]; B-operand from v^T.
__global__ void __launch_bounds__(256, 2)
pv_kernel(const u16* __restrict__ P, const u16* __restrict__ VT,
          float* __restrict__ out) {
  __shared__ u16 sp[128 * LSTR];
  __shared__ u16 sv[64 * LSTR];
  const int l0 = blockIdx.x * 128;
  const int bh = blockIdx.y;
  const int b = bh / 12, hh = bh % 12;
  const int tid = threadIdx.x;
  const int wave = tid >> 5;
  const unsigned lane = tid & 31u, sel = lane >> 4, jl = lane & 15u;
  const int i0 = wave * 16;

  v8f acc[4];
#pragma unroll
  for (int t = 0; t < 4; ++t) acc[t] = v8f{};

  for (int r0 = 0; r0 < 1024; r0 += 64) {
    load_tile64(sp, P + ((size_t)bh * 1024 + l0) * 1024 + r0, 1024, 128);
    load_tile64(sv, VT + (size_t)bh * 64 * 1024 + r0, 1024, 64);
    async_join();
    __syncthreads();
    v16bf a0 = load_a_frag(sp, i0, 0);
    v16bf a1 = load_a_frag(sp, i0, 1);
#pragma unroll
    for (int t = 0; t < 4; ++t) {
      v16bf b0 = load_b_frag(sv, t * 16, 0);
      v16bf b1 = load_b_frag(sv, t * 16, 1);
      acc[t] = wmma_bf16(a0, b0, acc[t]);
      acc[t] = wmma_bf16(a1, b1, acc[t]);
    }
    __syncthreads();
  }
#pragma unroll
  for (int t = 0; t < 4; ++t)
#pragma unroll
    for (int v = 0; v < 8; ++v) {
      int l = l0 + i0 + v + (int)sel * 8;
      int n = t * 16 + (int)jl;
      out[((size_t)b * 1024 + l) * 768 + hh * 64 + n] = acc[t][v];
    }
}

// ------------------------------ host launcher ------------------------------

extern "C" void kernel_launch(void* const* d_in, const int* in_sizes, int n_in,
                              void* d_out, int out_size, void* d_ws, size_t ws_size,
                              hipStream_t stream) {
  (void)in_sizes; (void)n_in; (void)out_size; (void)ws_size;
  const float* hidden  = (const float*)d_in[0];
  const float* mask    = (const float*)d_in[1];
  const float* structm = (const float*)d_in[2];
  const float* Wq = (const float*)d_in[3];
  const float* bq = (const float*)d_in[4];
  const float* Wk = (const float*)d_in[5];
  const float* bk = (const float*)d_in[6];
  const float* Wv = (const float*)d_in[7];
  const float* bv = (const float*)d_in[8];
  const float* de = (const float*)d_in[9];
  const float* ssw = (const float*)d_in[10];
  const float* ab  = (const float*)d_in[11];
  float* out = (float*)d_out;

  char* ws = (char*)d_ws;
  size_t off = 0;
  auto take = [&](size_t bytes) -> char* {
    char* p = ws + off;
    off = (off + bytes + 255) & ~(size_t)255;
    return p;
  };
  u16* hb  = (u16*)take(2048ull * 768 * 2);      // hidden bf16
  u16* wtq = (u16*)take(768ull * 768 * 2);       // Wq^T bf16
  u16* wtk = (u16*)take(768ull * 768 * 2);
  u16* wtv = (u16*)take(768ull * 768 * 2);
  u16* eb  = (u16*)take(2047ull * 64 * 2);       // dist_emb bf16
  u16* sst = (u16*)take(5ull * 12 * 64 * 64 * 2);// ssan_w^T bf16
  u16* qb  = (u16*)take(24ull * 1024 * 64 * 2);  // q  [BH,L,64]
  u16* kb  = (u16*)take(24ull * 1024 * 64 * 2);  // k
  u16* vb  = (u16*)take(24ull * 1024 * 64 * 2);  // v (head-major copy)
  u16* vtb = (u16*)take(24ull * 64 * 1024 * 2);  // v^T [BH,64,L]
  u16* qwb = (u16*)take(5ull * 24 * 1024 * 64 * 2); // qw_i
  float* sc = (float*)take(24ull * 1024 * 1024 * 4); // scores f32
  u16* pb  = (u16*)take(24ull * 1024 * 1024 * 2);    // probs bf16

  conv_f2b<<<6144, 256, 0, stream>>>(hidden, hb, 2048 * 768);
  conv_transpose<<<2304, 256, 0, stream>>>(Wq, wtq, 768, 768);
  conv_transpose<<<2304, 256, 0, stream>>>(Wk, wtk, 768, 768);
  conv_transpose<<<2304, 256, 0, stream>>>(Wv, wtv, 768, 768);
  conv_f2b<<<512, 256, 0, stream>>>(de, eb, 2047 * 64);
  conv_ssanT<<<960, 256, 0, stream>>>(ssw, sst);

  dim3 gg(16, 6);
  gemm_qkv<<<gg, 256, 0, stream>>>(hb, wtq, bq, qb, vtb, 0);
  gemm_qkv<<<gg, 256, 0, stream>>>(hb, wtk, bk, kb, vtb, 0);
  gemm_qkv<<<gg, 256, 0, stream>>>(hb, wtv, bv, vb, vtb, 1);

  gemm_qw<<<dim3(8, 120), 256, 0, stream>>>(qb, sst, qwb);

  // dynamic LDS: (2*128 + 256 + 5*128)*LSTR halves + 8*2*16*33 floats = 199680B
  scores_kernel<<<dim3(64, 24), 256, 199680, stream>>>(qb, kb, qwb, eb, structm,
                                                       mask, ab, sc);

  softmax_kernel<<<24576, 256, 0, stream>>>(sc, pb);
  pv_kernel<<<dim3(8, 24), 256, 0, stream>>>(pb, vtb, out);
}
